// HeteroGraphTransformer_68289980006598
// MI455X (gfx1250) — compile-verified, matrix operand order checked by reading, not run on hardware
//
#include <hip/hip_runtime.h>
#include <math.h>

// ---------------------------------------------------------------------------
// HGT (hetero graph transformer) for MI455X / gfx1250.
// Dense 32x32 projections via v_wmma_f32_16x16x32_f16 (K=32 == WMMA K),
// 4 row-tiles per wave (8 WMMAs) with the f16 weight tile held in registers.
// Irregular edge phase via VALU + global float/uint atomics (bandwidth-bound).
// N_ADDR and N_TX are multiples of 16 -> no per-row tail guards needed.
// ---------------------------------------------------------------------------

#define HGT_NA 300000
#define HGT_NT 400000
#define HGT_EIN 800000
#define HGT_EOUT 800000
#define HGT_ESP 500000

typedef __attribute__((ext_vector_type(16))) _Float16 v16h;
typedef __attribute__((ext_vector_type(8)))  float    v8f;

// ---------------- fill ----------------
__global__ void k_fill_u32(unsigned* __restrict__ p, unsigned val, long n) {
    long i = blockIdx.x * (long)blockDim.x + threadIdx.x;
    if (i < n) p[i] = val;
}

// ---------------- encoder: h = relu(x @ W + b), W: [F,32] ----------------
__global__ void k_encoder(const float* __restrict__ x, const float* __restrict__ W,
                          const float* __restrict__ b, float* __restrict__ h,
                          int n, int F) {
    long i = blockIdx.x * (long)blockDim.x + threadIdx.x;
    if (i >= (long)n * 32) return;
    long node = i >> 5;
    int  c    = (int)(i & 31);
    const float* xr = x + node * F;
    float acc = b[c];
    for (int f = 0; f < F; ++f) acc = fmaf(xr[f], W[f * 32 + c], acc);
    h[i] = fmaxf(acc, 0.f);
}

// ---------------- layernorm over 32 channels; lane == channel (wave32) ------
__global__ void k_layernorm(const float* __restrict__ h, const float* __restrict__ g,
                            const float* __restrict__ bb, float* __restrict__ xn,
                            _Float16* __restrict__ xnh, int n) {
    int  lane = threadIdx.x & 31;
    long node = (blockIdx.x * (long)blockDim.x + threadIdx.x) >> 5;
    if (node >= n) return;
    float v = h[node * 32 + lane];
    float s = v;
    #pragma unroll
    for (int m = 16; m >= 1; m >>= 1) s += __shfl_xor(s, m, 32);
    float mean = s * (1.f / 32.f);
    float dv   = v - mean;
    float q = dv * dv;
    #pragma unroll
    for (int m = 16; m >= 1; m >>= 1) q += __shfl_xor(q, m, 32);
    float var = q * (1.f / 32.f);
    float r   = rsqrtf(var + 1e-5f);
    float o   = dv * r * g[lane] + bb[lane];
    xn[node * 32 + lane]  = o;
    xnh[node * 32 + lane] = (_Float16)o;
}

// ---------------- WMMA GEMM helpers -----------------------------------------
// A layout (16-bit 16x32): lane<16 row=lane holds K 0..7 (v0..3) & 16..23 (v4..7);
// lane>=16 same row, K 8..15 & 24..31.  B mirrored with N in place of M.
// C layout: VGPR i, lane L -> row = i + (L>=16 ? 8 : 0), col = L&15.

__device__ __forceinline__ void load_b_tiles(const float* __restrict__ W, int lane,
                                             v16h& b0, v16h& b1) {
    int cl = lane & 15;
    int kb = (lane >> 4) * 8;
    #pragma unroll
    for (int j = 0; j < 8; ++j) {
        b0[j]     = (_Float16)W[(kb + j)      * 32 + cl];
        b0[j + 8] = (_Float16)W[(kb + 16 + j) * 32 + cl];
        b1[j]     = (_Float16)W[(kb + j)      * 32 + 16 + cl];
        b1[j + 8] = (_Float16)W[(kb + 16 + j) * 32 + 16 + cl];
    }
}

__device__ __forceinline__ void load_a_tile(const _Float16* __restrict__ A,
                                            long rowbase, int lane, v16h& a) {
    int cl = lane & 15;
    int kb = (lane >> 4) * 8;
    const _Float16* ar = A + (rowbase + cl) * 32 + kb;
    #pragma unroll
    for (int j = 0; j < 8; ++j) { a[j] = ar[j]; a[j + 8] = ar[16 + j]; }
}

// out = A @ W + bias   (K/Q/V projections); 4 tiles (64 rows) per wave
__global__ void k_gemm32_bias(const _Float16* __restrict__ A, const float* __restrict__ W,
                              const float* __restrict__ bias, float* __restrict__ out, int n) {
    int  lane = threadIdx.x & 31;
    int  wave = threadIdx.x >> 5;
    int  cl   = lane & 15;
    int  mo   = (lane >> 4) * 8;
    v16h b0, b1;
    load_b_tiles(W, lane, b0, b1);
    float bia0 = bias[cl], bia1 = bias[16 + cl];
    long tile0 = (blockIdx.x * 8L + wave) * 4;
    #pragma unroll
    for (int t = 0; t < 4; ++t) {
        long rowbase = (tile0 + t) * 16;
        if (rowbase < n) {                           // wave-uniform; n % 16 == 0
            v16h a;
            load_a_tile(A, rowbase, lane, a);
            v8f c0 = {}, c1 = {};
            c0 = __builtin_amdgcn_wmma_f32_16x16x32_f16(false, a, false, b0, (short)0, c0, false, false);
            c1 = __builtin_amdgcn_wmma_f32_16x16x32_f16(false, a, false, b1, (short)0, c1, false, false);
            float* o = out + (rowbase + mo) * 32 + cl;
            #pragma unroll
            for (int i = 0; i < 8; ++i) {
                o[i * 32]      = c0[i] + bia0;
                o[i * 32 + 16] = c1[i] + bia1;
            }
        }
    }
}

// o = A @ Wa + ba;  r = sig(skip)*o + (1-sig(skip))*xn;  h += relu(r)
__global__ void k_gemm32_out(const _Float16* __restrict__ A, const float* __restrict__ W,
                             const float* __restrict__ ba, const float* __restrict__ xn,
                             const float* __restrict__ skip, float* __restrict__ h, int n) {
    int  lane = threadIdx.x & 31;
    int  wave = threadIdx.x >> 5;
    int  cl   = lane & 15;
    int  mo   = (lane >> 4) * 8;
    v16h b0, b1;
    load_b_tiles(W, lane, b0, b1);
    float bia0 = ba[cl], bia1 = ba[16 + cl];
    float sa = 1.f / (1.f + expf(-skip[0]));
    long tile0 = (blockIdx.x * 8L + wave) * 4;
    #pragma unroll
    for (int t = 0; t < 4; ++t) {
        long rowbase = (tile0 + t) * 16;
        if (rowbase < n) {                           // wave-uniform; n % 16 == 0
            v16h a;
            load_a_tile(A, rowbase, lane, a);
            v8f c0 = {}, c1 = {};
            c0 = __builtin_amdgcn_wmma_f32_16x16x32_f16(false, a, false, b0, (short)0, c0, false, false);
            c1 = __builtin_amdgcn_wmma_f32_16x16x32_f16(false, a, false, b1, (short)0, c1, false, false);
            const float* xr = xn + (rowbase + mo) * 32 + cl;
            float*       hr = h  + (rowbase + mo) * 32 + cl;
            #pragma unroll
            for (int i = 0; i < 8; ++i) {
                float r0 = sa * (c0[i] + bia0) + (1.f - sa) * xr[i * 32];
                float r1 = sa * (c1[i] + bia1) + (1.f - sa) * xr[i * 32 + 16];
                hr[i * 32]      += fmaxf(r0, 0.f);
                hr[i * 32 + 16] += fmaxf(r1, 0.f);
            }
        }
    }
}

// ---------------- edge phase ----------------
__device__ __forceinline__ unsigned f2ord(float f) {
    unsigned b = __float_as_uint(f);
    return (b & 0x80000000u) ? ~b : (b | 0x80000000u);
}
__device__ __forceinline__ float ord2f(unsigned u) {
    unsigned b = (u & 0x80000000u) ? (u ^ 0x80000000u) : ~u;
    return __uint_as_float(b);
}

// score[e,h] = p_rel[h]/sqrt(8) * sum_f (sum_d K[src,h,d]*a_rel[h,d,f]) * Q[dst,h,f]
__global__ void k_edge_score(const int* __restrict__ src, const int* __restrict__ dst,
                             const float* __restrict__ K, const float* __restrict__ Q,
                             const float* __restrict__ a_rel, const float* __restrict__ p_rel,
                             float* __restrict__ sc, unsigned* __restrict__ mx, long E) {
    __shared__ float sA[256];
    __shared__ float sP[4];
    sA[threadIdx.x] = a_rel[threadIdx.x];                 // blockDim == 256
    if (threadIdx.x < 4) sP[threadIdx.x] = p_rel[threadIdx.x];
    __syncthreads();
    long e = blockIdx.x * (long)blockDim.x + threadIdx.x;
    if (e >= E) return;
    long s = src[e], d = dst[e];
    const float* kr = K + s * 32;
    const float* qr = Q + d * 32;
    const float scale = 0.3535533905932738f;              // 1/sqrt(D=8)
    #pragma unroll
    for (int hh = 0; hh < 4; ++hh) {
        float acc = 0.f;
        #pragma unroll
        for (int f = 0; f < 8; ++f) {
            float ke = 0.f;
            #pragma unroll
            for (int dd = 0; dd < 8; ++dd)
                ke = fmaf(kr[hh * 8 + dd], sA[hh * 64 + dd * 8 + f], ke);
            acc = fmaf(ke, qr[hh * 8 + f], acc);
        }
        float sv = acc * sP[hh] * scale;
        sc[e * 4 + hh] = sv;
        atomicMax(&mx[d * 4 + hh], f2ord(sv));
    }
}

// sc[e,h] = exp(sc - mx[dst,h]);  den[dst,h] += sc
__global__ void k_edge_exp(const int* __restrict__ dst, float* __restrict__ sc,
                           const unsigned* __restrict__ mx, float* __restrict__ den, long E) {
    long e = blockIdx.x * (long)blockDim.x + threadIdx.x;
    if (e >= E) return;
    long d = dst[e];
    #pragma unroll
    for (int hh = 0; hh < 4; ++hh) {
        float m  = ord2f(mx[d * 4 + hh]);
        float ev = expf(sc[e * 4 + hh] - m);
        sc[e * 4 + hh] = ev;
        atomicAdd(&den[d * 4 + hh], ev);
    }
}

// agg[dst, h*8+f] += alpha[e,h] * sum_d V[src,h,d]*m_rel[h,d,f]
__global__ void k_edge_agg(const int* __restrict__ src, const int* __restrict__ dst,
                           const float* __restrict__ V, const float* __restrict__ m_rel,
                           const float* __restrict__ sc, const float* __restrict__ den,
                           float* __restrict__ agg, long E) {
    __shared__ float sM[256];
    sM[threadIdx.x] = m_rel[threadIdx.x];
    __syncthreads();
    long e = blockIdx.x * (long)blockDim.x + threadIdx.x;
    if (e >= E) return;
    long s = src[e], d = dst[e];
    const float* vr = V + s * 32;
    #pragma unroll
    for (int hh = 0; hh < 4; ++hh) {
        float alpha = sc[e * 4 + hh] / (den[d * 4 + hh] + 1e-16f);
        #pragma unroll
        for (int f = 0; f < 8; ++f) {
            float ve = 0.f;
            #pragma unroll
            for (int dd = 0; dd < 8; ++dd)
                ve = fmaf(vr[hh * 8 + dd], sM[hh * 64 + dd * 8 + f], ve);
            atomicAdd(&agg[d * 32 + hh * 8 + f], ve * alpha);
        }
    }
}

// exact gelu -> f16 for the output WMMA GEMM
__global__ void k_gelu(const float* __restrict__ agg, _Float16* __restrict__ gh, long n) {
    long i = blockIdx.x * (long)blockDim.x + threadIdx.x;
    if (i >= n) return;
    float x = agg[i];
    gh[i] = (_Float16)(0.5f * x * (1.f + erff(x * 0.7071067811865476f)));
}

// final: out[n, 0..1] = h_addr[n,:] @ lin_W + lin_b
__global__ void k_final(const float* __restrict__ h, const float* __restrict__ W,
                        const float* __restrict__ b, float* __restrict__ out, int n) {
    long i = blockIdx.x * (long)blockDim.x + threadIdx.x;
    if (i >= n) return;
    const float* hr = h + i * 32;
    float a0 = b[0], a1 = b[1];
    #pragma unroll
    for (int c = 0; c < 32; ++c) {
        float v = hr[c];
        a0 = fmaf(v, W[c * 2 + 0], a0);
        a1 = fmaf(v, W[c * 2 + 1], a1);
    }
    out[i * 2 + 0] = a0;
    out[i * 2 + 1] = a1;
}

// ---------------------------------------------------------------------------
static inline long cdiv(long a, long b) { return (a + b - 1) / b; }

extern "C" void kernel_launch(void* const* d_in, const int* in_sizes, int n_in,
                              void* d_out, int out_size, void* d_ws, size_t ws_size,
                              hipStream_t stream) {
    (void)in_sizes; (void)n_in; (void)out_size; (void)ws_size;
    const int NA = HGT_NA, NT = HGT_NT;

    // ----- inputs (jax pytree flattening order: dict keys sorted, ASCII) -----
    const float* x_addr    = (const float*)d_in[0];
    const float* x_tx      = (const float*)d_in[1];
    const int*   ei_input  = (const int*)d_in[2];   // [2, E_IN]  addr -> tx
    const int*   ei_output = (const int*)d_in[3];   // [2, E_OUT] tx   -> addr
    const int*   ei_spent  = (const int*)d_in[4];   // [2, E_SP]  tx   -> tx
    const float* enc_addr_W = (const float*)d_in[5];
    const float* enc_addr_b = (const float*)d_in[6];
    const float* enc_tx_W   = (const float*)d_in[7];
    const float* enc_tx_b   = (const float*)d_in[8];
    // layers: base 9, 27 leaves each:
    // 0 Wa.addr 1 Wa.tx 2 Wk.addr 3 Wk.tx 4 Wq.addr 5 Wq.tx 6 Wv.addr 7 Wv.tx
    // 8 a_rel.in 9 a_rel.out 10 a_rel.sp 11 ba.addr 12 ba.tx 13 bk.addr 14 bk.tx
    // 15 bq.addr 16 bq.tx 17 bv.addr 18 bv.tx 19 m_rel.in 20 m_rel.out 21 m_rel.sp
    // 22 p_rel.in 23 p_rel.out 24 p_rel.sp 25 skip.addr 26 skip.tx
    auto LP = [&](int l, int off) -> const float* { return (const float*)d_in[9 + l * 27 + off]; };
    const float* lin_W     = (const float*)d_in[117];
    const float* lin_b     = (const float*)d_in[118];
    const float* ln_addr_b = (const float*)d_in[119];
    const float* ln_addr_g = (const float*)d_in[120];
    const float* ln_tx_b   = (const float*)d_in[121];
    const float* ln_tx_g   = (const float*)d_in[122];

    // ----- workspace bump allocator -----
    char*  ws  = (char*)d_ws;
    size_t off = 0;
    auto alloc = [&](size_t bytes) -> void* {
        void* p = ws + off;
        off = (off + bytes + 255) & ~(size_t)255;
        return p;
    };
    float*     h_addr = (float*)alloc((size_t)NA * 32 * 4);
    float*     h_tx   = (float*)alloc((size_t)NT * 32 * 4);
    float*     xn_a   = (float*)alloc((size_t)NA * 32 * 4);
    float*     xn_t   = (float*)alloc((size_t)NT * 32 * 4);
    _Float16*  xnh_a  = (_Float16*)alloc((size_t)NA * 32 * 2);
    _Float16*  xnh_t  = (_Float16*)alloc((size_t)NT * 32 * 2);
    float*     Ka = (float*)alloc((size_t)NA * 32 * 4);
    float*     Qa = (float*)alloc((size_t)NA * 32 * 4);
    float*     Va = (float*)alloc((size_t)NA * 32 * 4);
    float*     Kt = (float*)alloc((size_t)NT * 32 * 4);
    float*     Qt = (float*)alloc((size_t)NT * 32 * 4);
    float*     Vt = (float*)alloc((size_t)NT * 32 * 4);
    float*     agg = (float*)alloc((size_t)NT * 32 * 4);         // max(NA,NT) rows
    _Float16*  gh  = (_Float16*)alloc((size_t)NT * 32 * 2);
    float*     sc  = (float*)alloc((size_t)(HGT_EIN + HGT_ESP) * 4 * 4);
    unsigned*  mx  = (unsigned*)alloc((size_t)NT * 4 * 4);
    float*     den = (float*)alloc((size_t)NT * 4 * 4);

    const unsigned NEG_INF_ORD = 0x007FFFFFu;    // f2ord(-inf)
    const int B = 256;
    dim3 blk(B);
    dim3 gEA(cdiv((long)NA * 32, B)), gET(cdiv((long)NT * 32, B));
    dim3 gGA(cdiv(NA, 512)), gGT(cdiv(NT, 512));  // 8 waves/block, 64 rows/wave

    // ----- encoders -----
    k_encoder<<<gEA, blk, 0, stream>>>(x_addr, enc_addr_W, enc_addr_b, h_addr, NA, 53);
    k_encoder<<<gET, blk, 0, stream>>>(x_tx,   enc_tx_W,   enc_tx_b,   h_tx,   NT, 6);

    for (int l = 0; l < 4; ++l) {
        // layernorm (shared ln params across layers, per reference)
        k_layernorm<<<gEA, blk, 0, stream>>>(h_addr, ln_addr_g, ln_addr_b, xn_a, xnh_a, NA);
        k_layernorm<<<gET, blk, 0, stream>>>(h_tx,   ln_tx_g,   ln_tx_b,   xn_t, xnh_t, NT);

        // K/Q/V projections (WMMA)
        k_gemm32_bias<<<gGA, blk, 0, stream>>>(xnh_a, LP(l, 2), LP(l, 13), Ka, NA);
        k_gemm32_bias<<<gGA, blk, 0, stream>>>(xnh_a, LP(l, 4), LP(l, 15), Qa, NA);
        k_gemm32_bias<<<gGA, blk, 0, stream>>>(xnh_a, LP(l, 6), LP(l, 17), Va, NA);
        k_gemm32_bias<<<gGT, blk, 0, stream>>>(xnh_t, LP(l, 3), LP(l, 14), Kt, NT);
        k_gemm32_bias<<<gGT, blk, 0, stream>>>(xnh_t, LP(l, 5), LP(l, 16), Qt, NT);
        k_gemm32_bias<<<gGT, blk, 0, stream>>>(xnh_t, LP(l, 7), LP(l, 18), Vt, NT);

        // ===== dst = tx : relations 'input' (addr->tx) + 'spent' (tx->tx) =====
        k_fill_u32<<<cdiv((long)NT * 4, B), blk, 0, stream>>>(mx, NEG_INF_ORD, (long)NT * 4);
        k_fill_u32<<<cdiv((long)NT * 4, B), blk, 0, stream>>>((unsigned*)den, 0u, (long)NT * 4);
        k_fill_u32<<<cdiv((long)NT * 32, B), blk, 0, stream>>>((unsigned*)agg, 0u, (long)NT * 32);
        float* sc_sp = sc + (size_t)HGT_EIN * 4;
        k_edge_score<<<cdiv(HGT_EIN, B), blk, 0, stream>>>(ei_input, ei_input + HGT_EIN,
            Ka, Qt, LP(l, 8), LP(l, 22), sc, mx, HGT_EIN);
        k_edge_score<<<cdiv(HGT_ESP, B), blk, 0, stream>>>(ei_spent, ei_spent + HGT_ESP,
            Kt, Qt, LP(l, 10), LP(l, 24), sc_sp, mx, HGT_ESP);
        k_edge_exp<<<cdiv(HGT_EIN, B), blk, 0, stream>>>(ei_input + HGT_EIN, sc, mx, den, HGT_EIN);
        k_edge_exp<<<cdiv(HGT_ESP, B), blk, 0, stream>>>(ei_spent + HGT_ESP, sc_sp, mx, den, HGT_ESP);
        k_edge_agg<<<cdiv(HGT_EIN, B), blk, 0, stream>>>(ei_input, ei_input + HGT_EIN,
            Va, LP(l, 19), sc, den, agg, HGT_EIN);
        k_edge_agg<<<cdiv(HGT_ESP, B), blk, 0, stream>>>(ei_spent, ei_spent + HGT_ESP,
            Vt, LP(l, 21), sc_sp, den, agg, HGT_ESP);
        k_gelu<<<gET, blk, 0, stream>>>(agg, gh, (long)NT * 32);
        k_gemm32_out<<<gGT, blk, 0, stream>>>(gh, LP(l, 1), LP(l, 12), xn_t, LP(l, 26), h_tx, NT);

        // ===== dst = addr : relation 'output' (tx->addr) =====
        k_fill_u32<<<cdiv((long)NA * 4, B), blk, 0, stream>>>(mx, NEG_INF_ORD, (long)NA * 4);
        k_fill_u32<<<cdiv((long)NA * 4, B), blk, 0, stream>>>((unsigned*)den, 0u, (long)NA * 4);
        k_fill_u32<<<cdiv((long)NA * 32, B), blk, 0, stream>>>((unsigned*)agg, 0u, (long)NA * 32);
        k_edge_score<<<cdiv(HGT_EOUT, B), blk, 0, stream>>>(ei_output, ei_output + HGT_EOUT,
            Kt, Qa, LP(l, 9), LP(l, 23), sc, mx, HGT_EOUT);
        k_edge_exp<<<cdiv(HGT_EOUT, B), blk, 0, stream>>>(ei_output + HGT_EOUT, sc, mx, den, HGT_EOUT);
        k_edge_agg<<<cdiv(HGT_EOUT, B), blk, 0, stream>>>(ei_output, ei_output + HGT_EOUT,
            Vt, LP(l, 20), sc, den, agg, HGT_EOUT);
        k_gelu<<<gEA, blk, 0, stream>>>(agg, gh, (long)NA * 32);
        k_gemm32_out<<<gGA, blk, 0, stream>>>(gh, LP(l, 0), LP(l, 11), xn_a, LP(l, 25), h_addr, NA);
    }

    // ----- final linear -----
    k_final<<<cdiv(NA, B), blk, 0, stream>>>(h_addr, lin_W, lin_b, (float*)d_out, NA);
}
